// Up_23055384445334
// MI455X (gfx1250) — matrix operand order repaired and checked by reading, not compiled, not run on hardware
//
#include <hip/hip_runtime.h>
#include <hip/hip_bf16.h>

typedef __attribute__((ext_vector_type(16))) __bf16          v16bf;
typedef __attribute__((ext_vector_type(8)))  float           v8f;
typedef __attribute__((ext_vector_type(8)))  unsigned short  v8us;
typedef __attribute__((ext_vector_type(4)))  unsigned int    v4u;
typedef __attribute__((ext_vector_type(8)))  int             v8i;
typedef __attribute__((ext_vector_type(4)))  int             v4i;

#if __has_builtin(__builtin_amdgcn_tensor_load_to_lds) && __has_builtin(__builtin_amdgcn_s_wait_tensorcnt)
#define USE_TDM 1
#else
#define USE_TDM 0
#endif

// ---------------- bf16 helpers (round-to-nearest-even) ----------------
__device__ __forceinline__ unsigned short f2bf(float f) {
    union { float f; unsigned u; } v; v.f = f;
    unsigned r = v.u + 0x7FFFu + ((v.u >> 16) & 1u);
    return (unsigned short)(r >> 16);
}
__device__ __forceinline__ float bf2f(unsigned short h) {
    union { unsigned u; float f; } v; v.u = ((unsigned)h) << 16;
    return v.f;
}

// Problem constants
#define BATCH 2
#define CIN   512
#define COUT  256
#define HW    64
#define NPIX  4096            // 64*64
#define KDEF  4608            // 512*9
#define KTC   16              // 512/32
#define KTD   144             // 4608/32

#if USE_TDM
// Issue a TDM load of a 2-row tile (2 x 512 bf16 = 2KB) into LDS.
// Rows are the two M-tiles of this workgroup; row stride = KTD*512 elements.
// D# per CDNA5 ISA ch.8: group0 {count=1, lds_addr, global_addr, type=2},
// group1 {data_size=2B, tensor_dim0=512, tensor_dim1=2, tile_dim0=512,
//         tile_dim1=2, tensor_dim0_stride=KTD*512}.
__device__ __forceinline__ void tdm_load_a(const unsigned short* gptr, unsigned lds_off) {
    unsigned long long ga = (unsigned long long)(size_t)gptr;
    v4u g0;
    g0.x = 1u;                                                 // count=1, user mode
    g0.y = lds_off;                                            // LDS byte address
    g0.z = (unsigned)(ga & 0xFFFFFFFFu);                       // global_addr[31:0]
    g0.w = (unsigned)((ga >> 32) & 0x01FFFFFFu) | 0x80000000u; // addr[56:32] | type=2
    v8i g1;
    g1[0] = 0x00010000;                // workgroup_mask=0, data_size=1 (2 bytes)
    g1[1] = (int)(512u << 16);         // tensor_dim0[15:0] in bits[63:48]
    g1[2] = (int)(2u << 16);           // tensor_dim1[15:0] in bits[95:80]
    g1[3] = (int)(512u << 16);         // tile_dim0 in bits[127:112]
    g1[4] = 2;                         // tile_dim1 in bits[143:128]
    g1[5] = (int)(KTD * 512);          // tensor_dim0_stride low 32 bits
    g1[6] = 0;
    g1[7] = 0;
    v4i z4 = {0, 0, 0, 0};
    v8i z8 = {0, 0, 0, 0, 0, 0, 0, 0};
    __builtin_amdgcn_tensor_load_to_lds(g0, g1, z4, z4, z8, 0);
}
#endif

// ---------------------------------------------------------------------
// Pack fp32 row-major weights (M x K) into WMMA bf16 A-matrix lane layout.
// Tile (mt,kk) occupies 512 contiguous bf16: [(lane)*16 + e]
//   m = mt*16 + lane%16
//   k = kk*32 + e + (lane>=16 ? 16 : 0)
// ---------------------------------------------------------------------
__global__ void pack_a_bf16(const float* __restrict__ W, unsigned short* __restrict__ Apk,
                            int M, int K, int mtiles, int ktiles) {
    int idx = blockIdx.x * blockDim.x + threadIdx.x;
    int total = mtiles * ktiles * 512;
    if (idx >= total) return;
    int e    = idx & 15;
    int lane = (idx >> 4) & 31;
    int t    = idx >> 9;
    int kk   = t % ktiles;
    int mt   = t / ktiles;
    int m = mt * 16 + (lane & 15);
    int k = kk * 32 + e + ((lane >= 16) ? 16 : 0);
    float v = (m < M && k < K) ? W[(size_t)m * K + k] : 0.0f;
    Apk[idx] = f2bf(v);
}

// Bilinear upsample x1 (2,512,32,32) fp32 -> U (2,512,64,64) bf16
__global__ void upsample_bilinear(const float* __restrict__ x1, unsigned short* __restrict__ U) {
    int idx = blockIdx.x * blockDim.x + threadIdx.x;
    if (idx >= BATCH * CIN * NPIX) return;
    int x  = idx & 63;
    int y  = (idx >> 6) & 63;
    int bc = idx >> 12;
    const float s = 31.0f / 63.0f;
    float fy = y * s, fx = x * s;
    int i0 = min((int)fy, 30);
    int j0 = min((int)fx, 30);
    float wy = fy - (float)i0, wx = fx - (float)j0;
    const float* src = x1 + (size_t)bc * 1024;
    float v00 = src[i0 * 32 + j0],       v01 = src[i0 * 32 + j0 + 1];
    float v10 = src[(i0 + 1) * 32 + j0], v11 = src[(i0 + 1) * 32 + j0 + 1];
    float v = v00 * (1 - wy) * (1 - wx) + v01 * (1 - wy) * wx
            + v10 * wy * (1 - wx)       + v11 * wy * wx;
    U[idx] = f2bf(v);
}

// x2 (2,256,64,64) fp32 -> Xbf channels 0..255 (bf16)
__global__ void copy_x2(const float* __restrict__ x2, unsigned short* __restrict__ Xbf) {
    int idx = blockIdx.x * blockDim.x + threadIdx.x;
    if (idx >= BATCH * COUT * NPIX) return;
    int pix = idx & 4095;
    int c   = (idx >> 12) & 255;
    int b   = idx >> 20;
    Xbf[((size_t)(b * CIN + c) << 12) + pix] = f2bf(x2[idx]);
}

__global__ void zero_sums(float* __restrict__ s) {
    int i = blockIdx.x * blockDim.x + threadIdx.x;
    if (i < 512) s[i] = 0.0f;
}

// ---------------------------------------------------------------------
// 1x1 conv as WMMA GEMM: Out[m,n] = sum_k A[m,k]*In[k,n] + bias[m]
// In: bf16 [b][512][4096]. mode 0 -> bf16 into Xbf channels 256+m,
// mode 1 -> fp32 into OutF [b][256][4096].
// Block: 256 thr = 8 waves; tile M=32 x N=64; wave (mt,nt) owns 16x16.
// ---------------------------------------------------------------------
__global__ void __launch_bounds__(256)
conv1x1_wmma(const unsigned short* __restrict__ Apk, const unsigned short* __restrict__ In,
             const float* __restrict__ bias, unsigned short* __restrict__ OutBf,
             float* __restrict__ OutF, int mode) {
    __shared__ __align__(32) unsigned short Bs[2048]; // 4 subtiles * 512
    int lane = threadIdx.x & 31;
    int wave = threadIdx.x >> 5;
    int mt = wave >> 2, nt = wave & 3;
    int mbase = blockIdx.y * 32;
    int nbase = blockIdx.x * 64;
    int b    = nbase >> 12;
    int pix0 = nbase & 4095;
    const unsigned short* inB = In + ((size_t)b * CIN << 12);
    int mtileIdx = (mbase >> 4) + mt;
    const unsigned short* Aptr = Apk + (size_t)mtileIdx * KTC * 512 + lane * 16;

    v8f acc = {};
    for (int kk = 0; kk < KTC; ++kk) {
        __syncthreads();
        {   // stage B: 32 k-rows x 64 pixels; b128 global reads, swizzled b16 stores
            int t  = threadIdx.x;
            int k  = t >> 3;
            int n0 = (t & 7) * 8;
            const unsigned short* src = inB + ((size_t)(kk * 32 + k) << 12) + pix0 + n0;
#pragma unroll
            for (int j = 0; j < 8; ++j) {
                int n = n0 + j;
                Bs[(n >> 4) * 512 + ((n & 15) + (k & 16)) * 16 + (k & 15)] = src[j];
            }
        }
        __syncthreads();
        if (kk + 1 < KTC) __builtin_prefetch(Aptr + (kk + 1) * 512, 0, 0);
        v16bf a  = *(const v16bf*)(Aptr + kk * 512);
        v16bf bb = *(const v16bf*)(&Bs[nt * 512 + lane * 16]);
        acc = __builtin_amdgcn_wmma_f32_16x16x32_bf16(false, a, false, bb,
                                                      (short)0, acc, false, false);
    }
    int n = pix0 + nt * 16 + (lane & 15);
#pragma unroll
    for (int v = 0; v < 8; ++v) {
        int m = mbase + mt * 16 + ((lane >> 4) << 3) + v;
        float val = acc[v] + bias[m];
        if (mode == 0)
            OutBf[((size_t)(b * CIN + 256 + m) << 12) + n] = f2bf(val);
        else
            OutF[((size_t)(b * COUT + m) << 12) + n] = val;
    }
}

// ---------------------------------------------------------------------
// Offset 3x3 conv (18 out channels, padded to M=32) as WMMA GEMM with
// on-the-fly im2col: k = c*9 + (ky*3+kx).
// Staging: each thread owns (one n x 8 consecutive k) -> single ds_store_b128.
// ---------------------------------------------------------------------
__global__ void __launch_bounds__(256)
offconv_wmma(const unsigned short* __restrict__ Apk, const unsigned short* __restrict__ Xbf,
             const float* __restrict__ bias, float* __restrict__ Off) {
    __shared__ __align__(32) unsigned short Bs[2048];
    int lane = threadIdx.x & 31;
    int wave = threadIdx.x >> 5;
    int mt = wave >> 2, nt = wave & 3;
    int nbase = blockIdx.x * 64;
    int b    = nbase >> 12;
    int pix0 = nbase & 4095;
    int y    = pix0 >> 6;                 // tile = one image row, x = n
    const unsigned short* xb = Xbf + ((size_t)b * CIN << 12);
    const unsigned short* Aptr = Apk + (size_t)mt * KTD * 512 + lane * 16;

    v8f acc = {};
    for (int kk = 0; kk < KTD; ++kk) {
        __syncthreads();
        {
            int t     = threadIdx.x;
            int n     = t & 63;
            int khalf = t >> 6;           // 0..3, covers k = khalf*8 .. +7
            v8us tmp;
#pragma unroll
            for (int j = 0; j < 8; ++j) {
                int k   = khalf * 8 + j;
                int kg  = kk * 32 + k;
                int c   = kg / 9;
                int tap = kg - 9 * c;
                int ky = tap / 3 - 1;
                int kx = tap - (tap / 3) * 3 - 1;
                int yy = y + ky;
                int xx = n + kx;
                bool ok = (yy >= 0) && (yy < HW) && (xx >= 0) && (xx < HW);
                tmp[j] = ok ? xb[((size_t)c << 12) + (yy << 6) + xx] : (unsigned short)0;
            }
            int sl = (n & 15) + ((khalf >= 2) ? 16 : 0);
            *(v8us*)(&Bs[(n >> 4) * 512 + sl * 16 + (khalf & 1) * 8]) = tmp;
        }
        __syncthreads();
        if (kk + 1 < KTD) __builtin_prefetch(Aptr + (kk + 1) * 512, 0, 0);
        v16bf a  = *(const v16bf*)(Aptr + kk * 512);
        v16bf bb = *(const v16bf*)(&Bs[nt * 512 + lane * 16]);
        acc = __builtin_amdgcn_wmma_f32_16x16x32_bf16(false, a, false, bb,
                                                      (short)0, acc, false, false);
    }
    int n = nt * 16 + (lane & 15);
#pragma unroll
    for (int v = 0; v < 8; ++v) {
        int m = mt * 16 + ((lane >> 4) << 3) + v;
        if (m < 18)
            Off[((size_t)(b * 18 + m) << 12) + (y << 6) + n] = acc[v] + bias[m];
    }
}

// ---------------------------------------------------------------------
// Deformable-conv GEMM: Raw[o,n] = sum_{c,tap} w[o,c,tap]*bilinear(X[c], tap, n)
// - sampling geometry (9 taps x 4 corners) precomputed per 64-pixel row in LDS
// - A (packed weights) double-buffered in LDS via TDM tensor_load_to_lds
// - B staged as one ds_store_b128 per thread per K-step
// - epilogue: bias + raw store + shfl-reduced per-channel sum/sumsq atomics
// ---------------------------------------------------------------------
__global__ void __launch_bounds__(256)
deform_wmma(const unsigned short* __restrict__ Apk, const unsigned short* __restrict__ Xbf,
            const float* __restrict__ Off, const float* __restrict__ bias,
            float* __restrict__ Raw, float* __restrict__ sums) {
    __shared__ __align__(32) unsigned short Bs[2048];
    __shared__ __align__(32) unsigned short Aw[2][1024]; // 2 stages x (2 mtiles * 512)
    __shared__ int   sIdx[576][4];   // [pixel*9+tap][corner] flat y*64+x (clamped)
    __shared__ float sW[576][4];     // corner weights (0 if OOB)

    int lane = threadIdx.x & 31;
    int wave = threadIdx.x >> 5;
    int mt = wave >> 2, nt = wave & 3;
    int mbase = blockIdx.y * 32;
    int nbase = blockIdx.x * 64;
    int b    = nbase >> 12;
    int pix0 = nbase & 4095;
    int y    = pix0 >> 6;

    // ---- precompute sampling geometry for the 64-pixel row ----
    for (int t = threadIdx.x; t < 576; t += 256) {
        int tap  = t % 9;
        int xpix = t / 9;
        int ky = tap / 3 - 1;
        int kx = tap - (tap / 3) * 3 - 1;
        float dy = Off[((size_t)(b * 18 + tap * 2)     << 12) + (y << 6) + xpix];
        float dx = Off[((size_t)(b * 18 + tap * 2 + 1) << 12) + (y << 6) + xpix];
        float py = (float)y + (float)ky + dy;
        float px = (float)xpix + (float)kx + dx;
        float fy = floorf(py), fx = floorf(px);
        int y0 = (int)fy, x0 = (int)fx;
        float wy = py - fy, wx = px - fx;
#pragma unroll
        for (int q = 0; q < 4; ++q) {
            int yi = y0 + (q >> 1);
            int xi = x0 + (q & 1);
            float wgt = ((q >> 1) ? wy : 1.0f - wy) * ((q & 1) ? wx : 1.0f - wx);
            bool ok = (yi >= 0) && (yi < HW) && (xi >= 0) && (xi < HW);
            int yc = min(max(yi, 0), HW - 1);
            int xc = min(max(xi, 0), HW - 1);
            sIdx[t][q] = (yc << 6) + xc;
            sW[t][q]   = ok ? wgt : 0.0f;
        }
    }

    const unsigned short* Abase = Apk + (size_t)(blockIdx.y * 2) * KTD * 512;
    const unsigned short* xb    = Xbf + ((size_t)b * CIN << 12);

#if USE_TDM
    if (threadIdx.x == 0)
        tdm_load_a(Abase + (size_t)0 * 512, (unsigned)(size_t)(void*)&Aw[0][0]);
#endif
    __syncthreads();

    v8f acc = {};
    for (int kk = 0; kk < KTD; ++kk) {
        __syncthreads();
        {   // stage B: thread owns (one n x 8 consecutive k) -> one ds_store_b128
            int t     = threadIdx.x;
            int n     = t & 63;
            int khalf = t >> 6;
            v8us tmp;
#pragma unroll
            for (int j = 0; j < 8; ++j) {
                int k   = khalf * 8 + j;
                int kg  = kk * 32 + k;
                int c   = kg / 9;
                int tap = kg - 9 * c;
                int s   = n * 9 + tap;
                const unsigned short* xc = xb + ((size_t)c << 12);
                float v = sW[s][0] * bf2f(xc[sIdx[s][0]])
                        + sW[s][1] * bf2f(xc[sIdx[s][1]])
                        + sW[s][2] * bf2f(xc[sIdx[s][2]])
                        + sW[s][3] * bf2f(xc[sIdx[s][3]]);
                tmp[j] = f2bf(v);
            }
            int sl = (n & 15) + ((khalf >= 2) ? 16 : 0);
            *(v8us*)(&Bs[(n >> 4) * 512 + sl * 16 + (khalf & 1) * 8]) = tmp;
        }
#if USE_TDM
        if (threadIdx.x == 0) {
            if (kk + 1 < KTD) {
                tdm_load_a(Abase + (size_t)(kk + 1) * 512,
                           (unsigned)(size_t)(void*)&Aw[(kk + 1) & 1][0]);
                __builtin_amdgcn_s_wait_tensorcnt((short)1);
            } else {
                __builtin_amdgcn_s_wait_tensorcnt((short)0);
            }
        }
#endif
        __syncthreads();
#if USE_TDM
        v16bf a = *(const v16bf*)(&Aw[kk & 1][mt * 512 + lane * 16]);
#else
        v16bf a = *(const v16bf*)(Abase + ((size_t)mt * KTD + kk) * 512 + lane * 16);
#endif
        v16bf bb = *(const v16bf*)(&Bs[nt * 512 + lane * 16]);
        acc = __builtin_amdgcn_wmma_f32_16x16x32_bf16(false, a, false, bb,
                                                      (short)0, acc, false, false);
    }

    int n = pix0 + nt * 16 + (lane & 15);
#pragma unroll
    for (int v = 0; v < 8; ++v) {
        int m = mbase + mt * 16 + ((lane >> 4) << 3) + v;
        float val = acc[v] + bias[m];
        Raw[((size_t)(b * COUT + m) << 12) + n] = val;
        float s1 = val, s2 = val * val;
#pragma unroll
        for (int d = 1; d < 16; d <<= 1) {
            s1 += __shfl_xor(s1, d);
            s2 += __shfl_xor(s2, d);
        }
        if ((lane & 15) == 0) {
            atomicAdd(&sums[m], s1);
            atomicAdd(&sums[256 + m], s2);
        }
    }
}

// BN (batch stats) + shortcut add + ReLU
__global__ void finalize_kernel(const float* __restrict__ Raw, const float* __restrict__ Sc,
                                const float* __restrict__ sums, const float* __restrict__ gamma,
                                const float* __restrict__ beta, float* __restrict__ out) {
    int idx = blockIdx.x * blockDim.x + threadIdx.x;
    if (idx >= BATCH * COUT * NPIX) return;
    int m = (idx >> 12) & 255;
    const float invN = 1.0f / (float)(BATCH * NPIX);
    float mean = sums[m] * invN;
    float var  = sums[256 + m] * invN - mean * mean;
    float v = (Raw[idx] - mean) * rsqrtf(var + 1e-5f) * gamma[m] + beta[m] + Sc[idx];
    out[idx] = fmaxf(v, 0.0f);
}

// ---------------------------------------------------------------------
extern "C" void kernel_launch(void* const* d_in, const int* in_sizes, int n_in,
                              void* d_out, int out_size, void* d_ws, size_t ws_size,
                              hipStream_t stream) {
    const float* x1    = (const float*)d_in[0];
    const float* x2    = (const float*)d_in[1];
    const float* up_w  = (const float*)d_in[2];
    const float* up_b  = (const float*)d_in[3];
    const float* off_w = (const float*)d_in[4];
    const float* off_b = (const float*)d_in[5];
    const float* def_w = (const float*)d_in[6];
    const float* def_b = (const float*)d_in[7];
    const float* gamma = (const float*)d_in[8];
    const float* beta  = (const float*)d_in[9];
    const float* sc_w  = (const float*)d_in[10];
    const float* sc_b  = (const float*)d_in[11];
    float* out = (float*)d_out;

    char* ws = (char*)d_ws;
    size_t off = 0;
    auto alloc = [&](size_t bytes) -> void* {
        void* p = ws + off;
        off = (off + bytes + 255) & ~(size_t)255;
        return p;
    };
    unsigned short* U      = (unsigned short*)alloc((size_t)BATCH * CIN * NPIX * 2);
    unsigned short* Xbf    = (unsigned short*)alloc((size_t)BATCH * CIN * NPIX * 2);
    unsigned short* AupPk  = (unsigned short*)alloc((size_t)16 * KTC * 512 * 2);
    unsigned short* AscPk  = (unsigned short*)alloc((size_t)16 * KTC * 512 * 2);
    unsigned short* AdefPk = (unsigned short*)alloc((size_t)16 * KTD * 512 * 2);
    unsigned short* AoffPk = (unsigned short*)alloc((size_t)2  * KTD * 512 * 2);
    float* Offs = (float*)alloc((size_t)BATCH * 18  * NPIX * 4);
    float* Raw  = (float*)alloc((size_t)BATCH * COUT * NPIX * 4);
    float* Sc   = (float*)alloc((size_t)BATCH * COUT * NPIX * 4);
    float* Sums = (float*)alloc(512 * 4);

    // weight packing
    pack_a_bf16<<<(16 * KTC * 512) / 256, 256, 0, stream>>>(up_w,  AupPk,  256, 512,  16, KTC);
    pack_a_bf16<<<(16 * KTC * 512) / 256, 256, 0, stream>>>(sc_w,  AscPk,  256, 512,  16, KTC);
    pack_a_bf16<<<(16 * KTD * 512) / 256, 256, 0, stream>>>(def_w, AdefPk, 256, KDEF, 16, KTD);
    pack_a_bf16<<<(2  * KTD * 512) / 256, 256, 0, stream>>>(off_w, AoffPk, 18,  KDEF, 2,  KTD);

    // inputs -> bf16
    upsample_bilinear<<<(BATCH * CIN * NPIX) / 256, 256, 0, stream>>>(x1, U);
    copy_x2<<<(BATCH * COUT * NPIX) / 256, 256, 0, stream>>>(x2, Xbf);

    // up 1x1 conv -> concat channels 256..511
    conv1x1_wmma<<<dim3(128, 8), 256, 0, stream>>>(AupPk, U, up_b, Xbf, nullptr, 0);

    // offset 3x3 conv
    offconv_wmma<<<128, 256, 0, stream>>>(AoffPk, Xbf, off_b, Offs);

    // deform GEMM + BN stats
    zero_sums<<<2, 256, 0, stream>>>(Sums);
    deform_wmma<<<dim3(128, 8), 256, 0, stream>>>(AdefPk, Xbf, Offs, def_b, Raw, Sums);

    // shortcut 1x1 conv
    conv1x1_wmma<<<dim3(128, 8), 256, 0, stream>>>(AscPk, Xbf, sc_b, nullptr, Sc, 1);

    // BN + shortcut + ReLU
    finalize_kernel<<<(BATCH * COUT * NPIX) / 256, 256, 0, stream>>>(Raw, Sc, Sums, gamma, beta, out);
}
